// MarkovERG_90898687852733
// MI455X (gfx1250) — matrix operand order repaired and checked by reading, not compiled, not run on hardware
//
#include <hip/hip_runtime.h>
#include <math.h>

// ---------------------------------------------------------------------------
// MI455X (gfx1250) implementation: bf16 WMMA GEMMs + fused attention.
// - Weights pre-transposed/converted to bf16 once (88MB total -> L2-resident).
// - slots converted to bf16 once (halves the dominant HBM stream).
// - GEMM uses GLOBAL_LOAD_ASYNC_TO_LDS_B128 (ASYNCcnt) with double-buffered
//   LDS tiles: DMA for chunk i+1 overlaps WMMA on chunk i, 1 barrier/iter.
// ---------------------------------------------------------------------------

typedef __attribute__((ext_vector_type(8)))  __bf16 v8bf;
typedef __attribute__((ext_vector_type(16))) __bf16 v16bf;
typedef __attribute__((ext_vector_type(8)))  float  v8f;

__device__ __forceinline__ unsigned short f2bf(float f) {
  unsigned int u = __float_as_uint(f);
  u += 0x7FFFu + ((u >> 16) & 1u);          // round-to-nearest-even
  return (unsigned short)(u >> 16);
}
__device__ __forceinline__ float bf2f(unsigned short s) {
  return __uint_as_float(((unsigned int)s) << 16);
}

// 16B global -> LDS async DMA (GVS mode: sgpr base + vgpr byte offset).
__device__ __forceinline__ void async_cp16(unsigned int lds_addr,
                                           unsigned int gbyte_off,
                                           const void* gbase) {
  asm volatile("global_load_async_to_lds_b128 %0, %1, %2"
               :
               : "v"(lds_addr), "v"(gbyte_off), "s"(gbase)
               : "memory");
}
// 32B (two 16B) sharing one address pair; IOFFSET applies to BOTH LDS+global.
__device__ __forceinline__ void async_cp32(unsigned int lds_addr,
                                           unsigned int gbyte_off,
                                           const void* gbase) {
  asm volatile("global_load_async_to_lds_b128 %0, %1, %2\n\t"
               "global_load_async_to_lds_b128 %0, %1, %2 offset:16"
               :
               : "v"(lds_addr), "v"(gbyte_off), "s"(gbase)
               : "memory");
}
__device__ __forceinline__ void wait_async0() {
  asm volatile("s_wait_asynccnt 0x0" ::: "memory");
}

// ---------------------------------------------------------------------------
// Weight prep: fp32 (K x N) row-major  ->  bf16 (N x K) row-major (N-major)
// ---------------------------------------------------------------------------
__global__ __launch_bounds__(256) void transpose_to_bf16(
    const float* __restrict__ in, unsigned short* __restrict__ out, int K, int N) {
  __shared__ float tile[32][33];
  const int nb = blockIdx.x * 32;
  const int kb = blockIdx.y * 32;
  const int tx = threadIdx.x;     // 0..31
  const int ty = threadIdx.y;     // 0..7
#pragma unroll
  for (int i = 0; i < 32; i += 8)
    tile[ty + i][tx] = in[(size_t)(kb + ty + i) * N + nb + tx];
  __syncthreads();
#pragma unroll
  for (int i = 0; i < 32; i += 8)
    out[(size_t)(nb + ty + i) * K + kb + tx] = f2bf(tile[tx][ty + i]);
}

__global__ __launch_bounds__(256) void f32_to_bf16_kernel(
    const float* __restrict__ in, unsigned short* __restrict__ out, size_t n) {
  size_t i = (size_t)blockIdx.x * blockDim.x + threadIdx.x;
  const size_t stride = (size_t)gridDim.x * blockDim.x;
  for (; i < n; i += stride) out[i] = f2bf(in[i]);
}

// ---------------------------------------------------------------------------
// LayerNorm over D=2048, one row per block (256 threads, 8 elems/thread).
// ---------------------------------------------------------------------------
__global__ __launch_bounds__(256) void ln_kernel(
    const float* __restrict__ h, const float* __restrict__ g,
    const float* __restrict__ b, unsigned short* __restrict__ out_bf,
    unsigned short* __restrict__ raw_bf) {
  const int row = blockIdx.x;
  const int tid = threadIdx.x;
  const int lane = tid & 31, wave = tid >> 5;
  __shared__ float r1[8], r2[8];

  float x[8];
  float s1 = 0.f, s2 = 0.f;
#pragma unroll
  for (int k = 0; k < 8; ++k) {
    float v = h[(size_t)row * 2048 + tid + k * 256];
    x[k] = v; s1 += v; s2 += v * v;
  }
#pragma unroll
  for (int off = 16; off > 0; off >>= 1) {
    s1 += __shfl_xor(s1, off, 32);
    s2 += __shfl_xor(s2, off, 32);
  }
  if (lane == 0) { r1[wave] = s1; r2[wave] = s2; }
  __syncthreads();
  if (wave == 0) {
    float a = (lane < 8) ? r1[lane] : 0.f;
    float c = (lane < 8) ? r2[lane] : 0.f;
#pragma unroll
    for (int off = 4; off > 0; off >>= 1) {
      a += __shfl_xor(a, off, 32);
      c += __shfl_xor(c, off, 32);
    }
    if (lane == 0) { r1[0] = a; r2[0] = c; }
  }
  __syncthreads();
  const float mean = r1[0] * (1.0f / 2048.0f);
  const float var  = r2[0] * (1.0f / 2048.0f) - mean * mean;
  const float rstd = rsqrtf(var + 1e-5f);
#pragma unroll
  for (int k = 0; k < 8; ++k) {
    const int d = tid + k * 256;
    const float y = (x[k] - mean) * rstd * g[d] + b[d];
    out_bf[(size_t)row * 2048 + d] = f2bf(y);
    if (raw_bf) raw_bf[(size_t)row * 4096 + d] = f2bf(x[k]);
  }
}

// ---------------------------------------------------------------------------
// WMMA bf16 GEMM: C(M x N) = act(A(M x K) @ W'(N x K)^T + bias), fused
// epilogues. Workgroup tile 64x128, 8 waves, 32x32 per wave (2x2 WMMA).
// Double-buffered async-DMA staging (ASYNCcnt), one barrier per K-chunk.
//   EPI 0: bias,  bf16 out               (q projection)
//   EPI 1: bias + tanh-GELU, bf16 out    (FFN1)
//   EPI 2: sigmoid gate + h update, f32  (gate GEMM)
//   EPI 3: bias + residual add, f32      (FFN2)
// ---------------------------------------------------------------------------
template <int EPI>
__global__ __launch_bounds__(256) void gemm_wmma(
    const unsigned short* __restrict__ A,   // M x K bf16, stride lda
    const unsigned short* __restrict__ W,   // N x K bf16 (pre-transposed)
    const float* __restrict__ bias,         // N
    unsigned short* __restrict__ out_bf,    // EPI 0/1
    const float* hin,                       // EPI 2/3 (aliases hout)
    const float* __restrict__ ctx,          // EPI 2
    float* hout,                            // EPI 2/3
    int N, int K, int lda) {
  __shared__ alignas(16) unsigned short As[2][64 * 40];
  __shared__ alignas(16) unsigned short Bs[2][128 * 40];

  const int tid = threadIdx.x;
  const int lane = tid & 31;
  const int wave = tid >> 5;
  const int waveM = wave >> 2;                 // 0..1
  const int waveN = wave & 3;                  // 0..3
  const int m0 = blockIdx.y * 64;
  const int n0 = blockIdx.x * 128;

  const int arow = tid >> 2;                   // 0..63
  const int acol = (tid & 3) * 8;              // 0/8/16/24
  const int brow = tid >> 1;                   // 0..127
  const int bcol = (tid & 1) * 16;             // 0/16

  const int l15 = lane & 31 & 15;
  const int koffA = (lane < 16) ? 0 : 8;       // ISA A-frag K layout
  const int koffB = (lane < 16) ? 0 : 8;       // ISA B-frag K layout

  // Per-thread LDS destinations (32-bit LDS offsets) for both buffers.
  unsigned int as_dst[2], bs_dst[2];
#pragma unroll
  for (int bsel = 0; bsel < 2; ++bsel) {
    as_dst[bsel] = (unsigned int)(size_t)&As[bsel][arow * 40 + acol];
    bs_dst[bsel] = (unsigned int)(size_t)&Bs[bsel][brow * 40 + bcol];
  }
  // Per-thread global byte offsets at k0 = 0 (advance by 64B per chunk).
  unsigned int a_off = (unsigned int)(((size_t)(m0 + arow) * lda + acol) * 2);
  unsigned int b_off = (unsigned int)(((size_t)(n0 + brow) * K + bcol) * 2);

  v8f acc[2][2] = {};

  // Prologue: DMA chunk 0 into buffer 0.
  async_cp16(as_dst[0], a_off, A);
  async_cp32(bs_dst[0], b_off, W);

  const int niter = K >> 5;
  for (int it = 0; it < niter; ++it) {
    const int cur = it & 1;
    // Publish current buffer: my DMAs done (in-order), then everyone's.
    // The barrier also proves all waves finished reading buffer cur^1
    // (their fragment ds_loads were consumed by WMMA before arriving here),
    // so it is safe to overwrite it with the next chunk's DMA below.
    wait_async0();
    __syncthreads();
    if (it + 1 < niter) {
      const unsigned int knext = (unsigned int)((it + 1) << 6); // 32 elems * 2B
      async_cp16(as_dst[cur ^ 1], a_off + knext, A);
      async_cp32(bs_dst[cur ^ 1], b_off + knext, W);
    }

    v16bf aF[2], bF[2];
#pragma unroll
    for (int i = 0; i < 2; ++i) {
      union { v16bf v; v8bf h[2]; } fa, fb;
      const unsigned short* ap = &As[cur][(waveM * 32 + i * 16 + l15) * 40];
      fa.h[0] = *(const v8bf*)(ap + koffA);
      fa.h[1] = *(const v8bf*)(ap + 16 + koffA);
      aF[i] = fa.v;
      const unsigned short* bp = &Bs[cur][(waveN * 32 + i * 16 + l15) * 40 + koffB];
      fb.h[0] = *(const v8bf*)(bp);
      fb.h[1] = *(const v8bf*)(bp + 16);
      bF[i] = fb.v;
    }
#pragma unroll
    for (int i = 0; i < 2; ++i)
#pragma unroll
      for (int j = 0; j < 2; ++j)
        acc[i][j] = __builtin_amdgcn_wmma_f32_16x16x32_bf16(
            false, aF[i], false, bF[j], (short)0, acc[i][j], false, false);
  }

  const int mlane = (lane < 16) ? 0 : 8;       // ISA C-frag M layout
#pragma unroll
  for (int i = 0; i < 2; ++i) {
#pragma unroll
    for (int j = 0; j < 2; ++j) {
      const int n = n0 + waveN * 32 + j * 16 + l15;
      const float bn = bias[n];
#pragma unroll
      for (int r = 0; r < 8; ++r) {
        const int m = m0 + waveM * 32 + i * 16 + mlane + r;
        const size_t idx = (size_t)m * N + n;
        const float v = acc[i][j][r] + bn;
        if (EPI == 0) {
          out_bf[idx] = f2bf(v);
        } else if (EPI == 1) {
          const float t = v;
          const float gl = 0.5f * t *
              (1.0f + tanhf(0.7978845608028654f * (t + 0.044715f * t * t * t)));
          out_bf[idx] = f2bf(gl);
        } else if (EPI == 2) {
          const float gg = 1.0f / (1.0f + __expf(-v));
          const float hv = hin[idx];
          const float cv = ctx[idx];
          hout[idx] = hv * (1.0f - gg) + cv * gg;
        } else {
          hout[idx] = hin[idx] + v;
        }
      }
    }
  }
}

// ---------------------------------------------------------------------------
// Fused attention: one block per batch row b.
// Phase 1: logits[s] = q[b]·slots[b,s] (wave-cooperative, coalesced)
// Phase 2: softmax over S=256 in LDS
// Phase 3: context[d] = sum_s p[s]*slots[b,s,d] (second pass hits L2)
// ---------------------------------------------------------------------------
template <bool SB>
__global__ __launch_bounds__(256) void attend_kernel(
    const unsigned short* __restrict__ qbf,  // B x 2048 bf16
    const void* __restrict__ slots_p,        // B x 256 x 2048 (bf16 or f32)
    float scale,
    float* __restrict__ ctx_f32,             // B x 2048
    unsigned short* __restrict__ hc) {       // B x 4096, cols 2048.. written
  const int b = blockIdx.x;
  const int tid = threadIdx.x;
  const int lane = tid & 31, wave = tid >> 5;
  __shared__ float qs[2048];
  __shared__ float pvals[256];
  __shared__ float red[8];

  for (int i = tid; i < 2048; i += 256) qs[i] = bf2f(qbf[(size_t)b * 2048 + i]);
  __syncthreads();

  const unsigned short* sb = (const unsigned short*)slots_p;
  const float* sf = (const float*)slots_p;

  // logits: wave w handles s in [w*32, w*32+32)
  for (int si = 0; si < 32; ++si) {
    const int s = wave * 32 + si;
    const size_t base = ((size_t)b * 256 + s) * 2048;
    float acc = 0.f;
#pragma unroll 4
    for (int d = lane * 4; d < 2048; d += 128) {
      float f0, f1, f2, f3;
      if (SB) {
        const uint2 u = *(const uint2*)(sb + base + d);
        f0 = __uint_as_float(u.x << 16);
        f1 = __uint_as_float(u.x & 0xffff0000u);
        f2 = __uint_as_float(u.y << 16);
        f3 = __uint_as_float(u.y & 0xffff0000u);
      } else {
        const float4 u = *(const float4*)(sf + base + d);
        f0 = u.x; f1 = u.y; f2 = u.z; f3 = u.w;
      }
      acc += qs[d] * f0 + qs[d + 1] * f1 + qs[d + 2] * f2 + qs[d + 3] * f3;
    }
#pragma unroll
    for (int off = 16; off > 0; off >>= 1) acc += __shfl_xor(acc, off, 32);
    if (lane == 0) pvals[s] = acc * scale;
  }
  __syncthreads();

  // softmax over 256 values
  const float v = pvals[tid];
  float m = v;
#pragma unroll
  for (int off = 16; off > 0; off >>= 1) m = fmaxf(m, __shfl_xor(m, off, 32));
  if (lane == 0) red[wave] = m;
  __syncthreads();
  if (wave == 0) {
    float a = (lane < 8) ? red[lane] : -3.0e38f;
#pragma unroll
    for (int off = 4; off > 0; off >>= 1) a = fmaxf(a, __shfl_xor(a, off, 32));
    if (lane == 0) red[0] = a;
  }
  __syncthreads();
  const float mx = red[0];
  const float e = __expf(v - mx);
  float s = e;
#pragma unroll
  for (int off = 16; off > 0; off >>= 1) s += __shfl_xor(s, off, 32);
  __syncthreads();
  if (lane == 0) red[wave] = s;
  __syncthreads();
  if (wave == 0) {
    float a = (lane < 8) ? red[lane] : 0.f;
#pragma unroll
    for (int off = 4; off > 0; off >>= 1) a += __shfl_xor(a, off, 32);
    if (lane == 0) red[0] = a;
  }
  __syncthreads();
  pvals[tid] = e / red[0];
  __syncthreads();

  // context
  float accv[8] = {0.f, 0.f, 0.f, 0.f, 0.f, 0.f, 0.f, 0.f};
  for (int si = 0; si < 256; ++si) {
    const float p = pvals[si];
    const size_t base = ((size_t)b * 256 + si) * 2048;
#pragma unroll
    for (int k = 0; k < 8; ++k) {
      const int d = tid + k * 256;
      const float sv = SB ? bf2f(sb[base + d]) : sf[base + d];
      accv[k] += p * sv;
    }
  }
#pragma unroll
  for (int k = 0; k < 8; ++k) {
    const int d = tid + k * 256;
    ctx_f32[(size_t)b * 2048 + d] = accv[k];
    hc[(size_t)b * 4096 + 2048 + d] = f2bf(accv[k]);
  }
}

// ---------------------------------------------------------------------------
extern "C" void kernel_launch(void* const* d_in, const int* in_sizes, int n_in,
                              void* d_out, int out_size, void* d_ws, size_t ws_size,
                              hipStream_t stream) {
  const float* anchor = (const float*)d_in[0];
  const float* slots  = (const float*)d_in[1];
  const float* wt  = (const float*)d_in[2];
  const float* bt  = (const float*)d_in[3];
  const float* wg  = (const float*)d_in[4];
  const float* bg  = (const float*)d_in[5];
  const float* g1  = (const float*)d_in[6];
  const float* b1  = (const float*)d_in[7];
  const float* g2  = (const float*)d_in[8];
  const float* b2  = (const float*)d_in[9];
  const float* w1  = (const float*)d_in[10];
  const float* bf1 = (const float*)d_in[11];
  const float* w2  = (const float*)d_in[12];
  const float* bf2 = (const float*)d_in[13];
  (void)in_sizes; (void)n_in; (void)out_size;

  char* p = (char*)d_ws;
  size_t off = 0;
  auto alloc = [&](size_t bytes) -> void* {
    void* r = p + off;
    off = (off + bytes + 255) & ~(size_t)255;
    return r;
  };

  float*          h   = (float*)alloc((size_t)256 * 2048 * 4);
  unsigned short* hn  = (unsigned short*)alloc((size_t)256 * 2048 * 2);
  unsigned short* qb  = (unsigned short*)alloc((size_t)256 * 2048 * 2);
  unsigned short* hc  = (unsigned short*)alloc((size_t)256 * 4096 * 2);
  float*          ctx = (float*)alloc((size_t)256 * 2048 * 4);
  unsigned short* a1  = (unsigned short*)alloc((size_t)256 * 8192 * 2);
  unsigned short* wtT = (unsigned short*)alloc((size_t)2048 * 2048 * 2);
  unsigned short* wgT = (unsigned short*)alloc((size_t)2048 * 4096 * 2);
  unsigned short* w1T = (unsigned short*)alloc((size_t)8192 * 2048 * 2);
  unsigned short* w2T = (unsigned short*)alloc((size_t)2048 * 8192 * 2);
  unsigned short* slotsb = (unsigned short*)alloc((size_t)256 * 256 * 2048 * 2);
  const bool use_bf_slots = (ws_size >= off);

  hipMemcpyAsync(h, anchor, (size_t)256 * 2048 * 4, hipMemcpyDeviceToDevice, stream);

  const dim3 tb(32, 8);
  transpose_to_bf16<<<dim3(2048 / 32, 2048 / 32), tb, 0, stream>>>(wt, wtT, 2048, 2048);
  transpose_to_bf16<<<dim3(2048 / 32, 4096 / 32), tb, 0, stream>>>(wg, wgT, 4096, 2048);
  transpose_to_bf16<<<dim3(8192 / 32, 2048 / 32), tb, 0, stream>>>(w1, w1T, 2048, 8192);
  transpose_to_bf16<<<dim3(2048 / 32, 8192 / 32), tb, 0, stream>>>(w2, w2T, 8192, 2048);
  if (use_bf_slots)
    f32_to_bf16_kernel<<<8192, 256, 0, stream>>>(slots, slotsb, (size_t)256 * 256 * 2048);

  const float scale = 1.0f / sqrtf(2048.0f);

  for (int step = 0; step < 6; ++step) {
    ln_kernel<<<256, 256, 0, stream>>>(h, g1, b1, hn, hc);
    gemm_wmma<0><<<dim3(16, 4), 256, 0, stream>>>(hn, wtT, bt, qb, nullptr, nullptr,
                                                  nullptr, 2048, 2048, 2048);
    if (use_bf_slots)
      attend_kernel<true><<<256, 256, 0, stream>>>(qb, slotsb, scale, ctx, hc);
    else
      attend_kernel<false><<<256, 256, 0, stream>>>(qb, slots, scale, ctx, hc);
    gemm_wmma<2><<<dim3(16, 4), 256, 0, stream>>>(hc, wgT, bg, nullptr, h, ctx, h,
                                                  2048, 4096, 4096);
    ln_kernel<<<256, 256, 0, stream>>>(h, g2, b2, hn, nullptr);
    gemm_wmma<1><<<dim3(64, 4), 256, 0, stream>>>(hn, w1T, bf1, a1, nullptr, nullptr,
                                                  nullptr, 8192, 2048, 2048);
    gemm_wmma<3><<<dim3(16, 4), 256, 0, stream>>>(a1, w2T, bf2, nullptr, h, nullptr, h,
                                                  2048, 8192, 8192);
  }

  hipMemcpyAsync(d_out, h, (size_t)256 * 2048 * 4, hipMemcpyDeviceToDevice, stream);
}
